// CapsuleNet_3D_73547019976799
// MI455X (gfx1250) — matrix operand order repaired and checked by reading, not compiled, not run on hardware
//
#include <hip/hip_runtime.h>
#include <hip/hip_bf16.h>
#include <math.h>

typedef __attribute__((ext_vector_type(16))) _Float16 v16h;
typedef __attribute__((ext_vector_type(4)))  _Float16 v4h;
typedef __attribute__((ext_vector_type(8)))  float    v8f;
typedef __attribute__((ext_vector_type(2)))  float    v2f;

#define BATCH 16
#define C1    128
#define SD    27          // conv1 output spatial
#define RTOT  16000       // 16*1000
#define KTOT  93312       // 128*729
#define KCH   2916        // KTOT/32
#define NT    8           // 128/16
#define MT    32          // M rows per block in pcaps GEMM
#define CH    2           // K-chunks per round
#define ROUNDS (KCH / CH) // 1458

// ---------------------------------------------------------------------------
// conv1 (3x3x3, stride1, valid) + ReLU, output stored as f16 for WMMA GEMM
// ---------------------------------------------------------------------------
__global__ void conv1_relu_half(const float* __restrict__ x,
                                const float* __restrict__ w,
                                const float* __restrict__ bias,
                                _Float16* __restrict__ h)
{
    long o = (long)blockIdx.x * blockDim.x + threadIdx.x;
    const long total = (long)BATCH * C1 * SD * SD * SD;
    if (o >= total) return;
    long t = o;
    int xw = (int)(t % 27); t /= 27;
    int yw = (int)(t % 27); t /= 27;
    int zw = (int)(t % 27); t /= 27;
    int c  = (int)(t % 128);
    int b  = (int)(t / 128);
    const float* xb = x + (long)b * 29 * 29 * 29;
    const float* wc = w + (long)c * 27;
    float s = bias[c];
#pragma unroll
    for (int dz = 0; dz < 3; ++dz)
#pragma unroll
        for (int dy = 0; dy < 3; ++dy)
#pragma unroll
            for (int dx = 0; dx < 3; ++dx)
                s += xb[((long)(zw + dz) * 29 + (yw + dy)) * 29 + (xw + dx)] *
                     wc[(dz * 3 + dy) * 3 + dx];
    s = fmaxf(s, 0.f);
    h[o] = (_Float16)s;
}

// ---------------------------------------------------------------------------
// im2col offset table in BYTES: k -> byte offset of element within h (f16)
// ---------------------------------------------------------------------------
__global__ void build_off(int* __restrict__ off)
{
    int k = blockIdx.x * blockDim.x + threadIdx.x;
    if (k >= KTOT) return;
    int ic = k / 729, rem = k % 729;
    int kd = rem / 81, kh = (rem / 9) % 9, kw = rem % 9;
    off[k] = (ic * 19683 + kd * 729 + kh * 27 + kw) * 2;
}

// ---------------------------------------------------------------------------
// Pre-pack pcaps weights (oc-major f32) into f16 WMMA B-lane layout, contiguous k order:
// bp[nt][kc][lane][j], B fragment K = (lane/16)*16 + j, N = nt*16 + lane%16
// ---------------------------------------------------------------------------
__global__ void pack_B(const float* __restrict__ w, _Float16* __restrict__ bp)
{
    long o = (long)blockIdx.x * blockDim.x + threadIdx.x;
    const long total = (long)NT * KCH * 32 * 16;
    if (o >= total) return;
    int j    = (int)(o & 15);
    int lane = (int)((o >> 4) & 31);
    int kc   = (int)((o >> 9) % KCH);
    int nt   = (int)(o / (512L * KCH));
    int g    = lane >> 4;
    int n    = nt * 16 + (lane & 15);
    int K    = g * 16 + j;
    long k   = (long)kc * 32 + K;
    bp[o] = (_Float16)w[(long)n * KTOT + k];
}

// ---------------------------------------------------------------------------
// pcaps implicit GEMM: M=16000, N=128, K=93312
// Block: 256 thr, M-tile=32 (2 sub-tiles), 8 waves each own a 16-col N tile.
// Double-buffered LDS staging, 2 K-chunks/round, 4 WMMA per wave per round.
// Gather for round r+1 issues before the WMMAs of round r (latency hiding).
// ---------------------------------------------------------------------------
__global__ void __launch_bounds__(256)
pcaps_gemm(const _Float16* __restrict__ h,
           const _Float16* __restrict__ bp,
           const int* __restrict__ offB,
           const float* __restrict__ pb,
           float* __restrict__ P)
{
    // [buf][chunk][msub][lane*16+j]
    __shared__ __align__(16) _Float16 a_lds[2][CH][2][32 * 16];
    const char* h8 = (const char*)h;
    int tid  = threadIdx.x;
    int wave = tid >> 5, lane = tid & 31;
    int m0   = blockIdx.x * MT;

    // ---- staging role: 4 consecutive K for (msub_s, lane_s), both chunks ----
    int msub_s = tid >> 7;                // 0..1
    int u      = tid & 127;
    int lane_s = u >> 2;                  // 0..31
    int j0     = (u & 3) * 4;             // 0,4,8,12
    int g_s    = lane_s >> 4;
    int M_s    = (lane_s & 15) + msub_s * 16;
    int Kb     = j0 + g_s * 16;  // first of 4 consecutive K
    int m_s  = m0 + M_s;
    int b_s  = m_s / 1000, sp_s = m_s % 1000;
    int od = sp_s / 100, oh = (sp_s / 10) % 10, ow = sp_s % 10;
    unsigned baseMB = (unsigned)((b_s * 128 * 19683 +
                                  (2 * od) * 729 + (2 * oh) * 27 + (2 * ow)) * 2);
    const int* offp = offB + Kb;
    int sidx = lane_s * 16 + j0;

    // ---- compute role ----
    const _Float16* bptr = bp + (((long)wave * KCH) * 32 + lane) * 16;
    v8f acc0 = {0.f, 0.f, 0.f, 0.f, 0.f, 0.f, 0.f, 0.f};
    v8f acc1 = {0.f, 0.f, 0.f, 0.f, 0.f, 0.f, 0.f, 0.f};

    v4h ra, rb;
    {   // prologue gather: round 0
        const int* op = offp;
        int4 oa = *(const int4*)(op);
        int4 ob = *(const int4*)(op + 32);
        ra[0] = *(const _Float16*)(h8 + baseMB + (unsigned)oa.x);
        ra[1] = *(const _Float16*)(h8 + baseMB + (unsigned)oa.y);
        ra[2] = *(const _Float16*)(h8 + baseMB + (unsigned)oa.z);
        ra[3] = *(const _Float16*)(h8 + baseMB + (unsigned)oa.w);
        rb[0] = *(const _Float16*)(h8 + baseMB + (unsigned)ob.x);
        rb[1] = *(const _Float16*)(h8 + baseMB + (unsigned)ob.y);
        rb[2] = *(const _Float16*)(h8 + baseMB + (unsigned)ob.z);
        rb[3] = *(const _Float16*)(h8 + baseMB + (unsigned)ob.w);
    }
    *(v4h*)&a_lds[0][0][msub_s][sidx] = ra;
    *(v4h*)&a_lds[0][1][msub_s][sidx] = rb;

    int p = 0;
    for (int r = 0; r < ROUNDS; ++r) {
        __syncthreads();
        bool more = (r + 1 < ROUNDS);
        if (more) {   // issue gather for next round before the WMMA burst
            const int* op = offp + (r + 1) * (CH * 32);
            int4 oa = *(const int4*)(op);
            int4 ob = *(const int4*)(op + 32);
            ra[0] = *(const _Float16*)(h8 + baseMB + (unsigned)oa.x);
            ra[1] = *(const _Float16*)(h8 + baseMB + (unsigned)oa.y);
            ra[2] = *(const _Float16*)(h8 + baseMB + (unsigned)oa.z);
            ra[3] = *(const _Float16*)(h8 + baseMB + (unsigned)oa.w);
            rb[0] = *(const _Float16*)(h8 + baseMB + (unsigned)ob.x);
            rb[1] = *(const _Float16*)(h8 + baseMB + (unsigned)ob.y);
            rb[2] = *(const _Float16*)(h8 + baseMB + (unsigned)ob.z);
            rb[3] = *(const _Float16*)(h8 + baseMB + (unsigned)ob.w);
        }
#pragma unroll
        for (int c = 0; c < CH; ++c) {
            v16h bfrag = *(const v16h*)bptr;
            bptr += 512;
            v16h aA = *(const v16h*)&a_lds[p][c][0][lane * 16];
            acc0 = __builtin_amdgcn_wmma_f32_16x16x32_f16(
                       false, aA, false, bfrag, (short)0, acc0, false, false);
            v16h aB = *(const v16h*)&a_lds[p][c][1][lane * 16];
            acc1 = __builtin_amdgcn_wmma_f32_16x16x32_f16(
                       false, aB, false, bfrag, (short)0, acc1, false, false);
        }
        if (more) {
            __builtin_prefetch(bptr + 2048, 0, 3);
            *(v4h*)&a_lds[p ^ 1][0][msub_s][sidx] = ra;
            *(v4h*)&a_lds[p ^ 1][1][msub_s][sidx] = rb;
        }
        p ^= 1;
    }

    int g  = lane >> 4;
    int oc = wave * 16 + (lane & 15);
    float biasv = pb[oc];
#pragma unroll
    for (int r8 = 0; r8 < 8; ++r8) {
        int m  = m0 + r8 + g * 8;
        int b2 = m / 1000, sp2 = m % 1000;
        P[((long)b2 * 128 + oc) * 1000 + sp2] = acc0[r8] + biasv;
        int m2 = m + 16;
        int b3 = m2 / 1000, sp3 = m2 % 1000;
        P[((long)b3 * 128 + oc) * 1000 + sp3] = acc1[r8] + biasv;
    }
}

// ---------------------------------------------------------------------------
// u = squash(reshape/transpose(P)) over the 8 primary-capsule dims
// ---------------------------------------------------------------------------
__global__ void squash_u(const float* __restrict__ P, float* __restrict__ u)
{
    int t = blockIdx.x * blockDim.x + threadIdx.x;
    if (t >= BATCH * RTOT) return;
    int r = t % RTOT, b = t / RTOT;
    int rc = r / 1000, sp = r % 1000;
    float v[8]; float n2 = 0.f;
#pragma unroll
    for (int i = 0; i < 8; ++i) {
        v[i] = P[((long)b * 128 + i * 16 + rc) * 1000 + sp];
        n2 += v[i] * v[i];
    }
    float n = sqrtf(n2);
    float scale = n2 / (1.f + n2) / (n + 1e-8f);
#pragma unroll
    for (int i = 0; i < 8; ++i)
        u[((long)b * RTOT + r) * 8 + i] = scale * v[i];
}

// ---------------------------------------------------------------------------
// priors[b,c,r,o] = sum_i u[b,r,i] * route_w[c,r,i,o]
// ---------------------------------------------------------------------------
__global__ void priors_k(const float* __restrict__ u, const float* __restrict__ rw,
                         float* __restrict__ pr)
{
    int t = blockIdx.x * blockDim.x + threadIdx.x;
    if (t >= 2 * RTOT * 16) return;
    int o = t & 15;
    int r = (t >> 4) % RTOT;
    int c = t / (16 * RTOT);
    float w[8];
#pragma unroll
    for (int i = 0; i < 8; ++i)
        w[i] = rw[(((long)c * RTOT + r) * 8 + i) * 16 + o];
    for (int b = 0; b < BATCH; ++b) {
        float s = 0.f;
#pragma unroll
        for (int i = 0; i < 8; ++i)
            s += u[((long)b * RTOT + r) * 8 + i] * w[i];
        pr[(((long)b * 2 + c) * RTOT + r) * 16 + o] = s;
    }
}

// ---------------------------------------------------------------------------
// dynamic routing, 3 iterations, one block per batch element
// ---------------------------------------------------------------------------
__global__ void __launch_bounds__(256)
routing(const float* __restrict__ pr, float* __restrict__ logits,
        float* __restrict__ caps)
{
    __shared__ float s_sum[32];
    __shared__ float s_v[32];
    int b = blockIdx.x, tid = threadIdx.x;
    float* L        = logits + (long)b * 2 * RTOT;
    const float* PR = pr + (long)b * 2 * RTOT * 16;
    for (int r = tid; r < RTOT; r += 256) { L[r] = 0.f; L[RTOT + r] = 0.f; }
    __syncthreads();
    for (int it = 0; it < 3; ++it) {
        if (tid < 32) s_sum[tid] = 0.f;
        __syncthreads();
        float part[32];
#pragma unroll
        for (int q = 0; q < 32; ++q) part[q] = 0.f;
        for (int r = tid; r < RTOT; r += 256) {
            float l0 = L[r], l1 = L[RTOT + r];
            float mx = fmaxf(l0, l1);
            float e0 = __expf(l0 - mx), e1 = __expf(l1 - mx);
            float inv = 1.f / (e0 + e1);
            float p0 = e0 * inv, p1 = e1 * inv;
            const float* p0p = PR + (long)r * 16;
            const float* p1p = PR + (long)RTOT * 16 + (long)r * 16;
#pragma unroll
            for (int o = 0; o < 16; ++o) {
                part[o]      += p0 * p0p[o];
                part[16 + o] += p1 * p1p[o];
            }
        }
#pragma unroll
        for (int q = 0; q < 32; ++q) atomicAdd(&s_sum[q], part[q]);
        __syncthreads();
        if (tid < 2) {
            float n2 = 0.f;
            for (int o = 0; o < 16; ++o) { float z = s_sum[tid * 16 + o]; n2 += z * z; }
            float n = sqrtf(n2);
            float sc = n2 / (1.f + n2) / (n + 1e-8f);
            for (int o = 0; o < 16; ++o) s_v[tid * 16 + o] = sc * s_sum[tid * 16 + o];
        }
        __syncthreads();
        if (it < 2) {
            for (int r = tid; r < RTOT; r += 256) {
                const float* p0p = PR + (long)r * 16;
                const float* p1p = PR + (long)RTOT * 16 + (long)r * 16;
                float d0 = 0.f, d1 = 0.f;
#pragma unroll
                for (int o = 0; o < 16; ++o) {
                    d0 += p0p[o] * s_v[o];
                    d1 += p1p[o] * s_v[16 + o];
                }
                L[r] += d0; L[RTOT + r] += d1;
            }
            __syncthreads();
        } else {
            if (tid < 32) caps[b * 32 + tid] = s_v[tid];
        }
    }
}

// ---------------------------------------------------------------------------
// classes = ||caps||, one-hot(argmax) mask -> decoder input d0 (16x32)
// ---------------------------------------------------------------------------
__global__ void classes_mask(const float* __restrict__ caps,
                             float* __restrict__ classes_out,
                             float* __restrict__ d0)
{
    __shared__ float cls[32];
    int tid = threadIdx.x;            // 32 threads
    int b = tid >> 1, c = tid & 1;
    float n2 = 0.f;
    for (int o = 0; o < 16; ++o) { float z = caps[b * 32 + c * 16 + o]; n2 += z * z; }
    float n = sqrtf(n2);
    cls[b * 2 + c] = n;
    classes_out[b * 2 + c] = n;
    __syncthreads();
    if (tid < 16) {
        int bb = tid;
        int cbest = (cls[bb * 2 + 1] > cls[bb * 2 + 0]) ? 1 : 0;
        for (int j = 0; j < 32; ++j)
            d0[bb * 32 + j] = ((j >> 4) == cbest) ? caps[bb * 32 + j] : 0.f;
    }
}

// ---------------------------------------------------------------------------
// Decoder FC layer via V_WMMA_F32_16X16X4_F32 (M=16 == batch), f32 weights
// streamed once (bandwidth bound). act: 0 = relu, 1 = sigmoid
// ---------------------------------------------------------------------------
__global__ void __launch_bounds__(128)
fc_wmma_f32(const float* __restrict__ A, const float* __restrict__ W,
            const float* __restrict__ bias, float* __restrict__ out,
            int K, int N, int act)
{
    int wave = threadIdx.x >> 5, lane = threadIdx.x & 31;
    int n0 = (blockIdx.x * 4 + wave) * 16;
    if (n0 >= N) return;
    int g  = lane >> 4;
    int n  = n0 + (lane & 15);
    bool nok = n < N;
    const float* arow = A + (long)(lane & 15) * K;   // M = lane%16
    v8f acc = {0.f, 0.f, 0.f, 0.f, 0.f, 0.f, 0.f, 0.f};
#pragma unroll 4
    for (int k0 = 0; k0 < K; k0 += 4) {
        v2f a, bm;
        a[0]  = arow[k0 + g * 2];
        a[1]  = arow[k0 + g * 2 + 1];
        bm[0] = nok ? W[(long)(k0 + g * 2) * N + n]     : 0.f;
        bm[1] = nok ? W[(long)(k0 + g * 2 + 1) * N + n] : 0.f;
        acc = __builtin_amdgcn_wmma_f32_16x16x4_f32(
                  false, a, false, bm, (short)0, acc, false, false);
    }
    if (!nok) return;
    float bv = bias[n];
#pragma unroll
    for (int r = 0; r < 8; ++r) {
        int M = r + g * 8;
        float v = acc[r] + bv;
        v = (act == 0) ? fmaxf(v, 0.f) : 1.f / (1.f + __expf(-v));
        out[(long)M * N + n] = v;
    }
}

// ---------------------------------------------------------------------------
extern "C" void kernel_launch(void* const* d_in, const int* in_sizes, int n_in,
                              void* d_out, int out_size, void* d_ws, size_t ws_size,
                              hipStream_t stream)
{
    (void)in_sizes; (void)n_in; (void)out_size; (void)ws_size;
    const float* x       = (const float*)d_in[0];
    const float* conv1_w = (const float*)d_in[1];
    const float* conv1_b = (const float*)d_in[2];
    const float* pcaps_w = (const float*)d_in[3];
    const float* pcaps_b = (const float*)d_in[4];
    const float* route_w = (const float*)d_in[5];
    const float* dec_w1  = (const float*)d_in[6];
    const float* dec_b1  = (const float*)d_in[7];
    const float* dec_w2  = (const float*)d_in[8];
    const float* dec_b2  = (const float*)d_in[9];
    const float* dec_w3  = (const float*)d_in[10];
    const float* dec_b3  = (const float*)d_in[11];
    const float* dec_w4  = (const float*)d_in[12];
    const float* dec_b4  = (const float*)d_in[13];
    float* out_f = (float*)d_out;      // [0..31] classes, [32..] recon

    char* ws = (char*)d_ws;
    size_t cur = 0;
    auto alloc = [&](size_t bytes) -> void* {
        void* p = ws + cur;
        cur = (cur + bytes + 255) & ~(size_t)255;
        return p;
    };
    _Float16* h_half = (_Float16*)alloc((size_t)40310784 * 2);       // conv1 out f16
    int*      offt   = (int*)     alloc((size_t)KTOT * 4);
    _Float16* bpack  = (_Float16*)alloc((size_t)NT * KCH * 512 * 2); // packed B
    float*    P      = (float*)   alloc((size_t)16 * 128 * 1000 * 4);
    float*    u      = (float*)   alloc((size_t)16 * RTOT * 8 * 4);
    float*    pri    = (float*)   alloc((size_t)16 * 2 * RTOT * 16 * 4);
    float*    logits = (float*)   alloc((size_t)16 * 2 * RTOT * 4);
    float*    caps   = (float*)   alloc(16 * 32 * 4);
    float*    d0     = (float*)   alloc(16 * 32 * 4);
    float*    a1     = (float*)   alloc(16 * 512 * 4);
    float*    a2     = (float*)   alloc(16 * 2048 * 4);
    float*    a3     = (float*)   alloc((size_t)16 * 10240 * 4);

    conv1_relu_half<<<157464, 256, 0, stream>>>(x, conv1_w, conv1_b, h_half);
    build_off<<<(KTOT + 255) / 256, 256, 0, stream>>>(offt);
    pack_B<<<46656, 256, 0, stream>>>(pcaps_w, bpack);
    pcaps_gemm<<<500, 256, 0, stream>>>(h_half, bpack, offt, pcaps_b, P);
    squash_u<<<(BATCH * RTOT + 255) / 256, 256, 0, stream>>>(P, u);
    priors_k<<<(2 * RTOT * 16 + 255) / 256, 256, 0, stream>>>(u, route_w, pri);
    routing<<<16, 256, 0, stream>>>(pri, logits, caps);
    classes_mask<<<1, 32, 0, stream>>>(caps, out_f, d0);
    fc_wmma_f32<<<(512 + 63) / 64, 128, 0, stream>>>(d0, dec_w1, dec_b1, a1, 32, 512, 0);
    fc_wmma_f32<<<(2048 + 63) / 64, 128, 0, stream>>>(a1, dec_w2, dec_b2, a2, 512, 2048, 0);
    fc_wmma_f32<<<(10240 + 63) / 64, 128, 0, stream>>>(a2, dec_w3, dec_b3, a3, 2048, 10240, 0);
    fc_wmma_f32<<<(27000 + 63) / 64, 128, 0, stream>>>(a3, dec_w4, dec_b4, out_f + 32, 10240, 27000, 1);
}
